// CRNN_Michael_79190607004262
// MI455X (gfx1250) — matrix operand order repaired and checked by reading, not compiled, not run on hardware
//
#include <hip/hip_runtime.h>
#include <math.h>

// ---------------------------------------------------------------------------
// CRNN forward for MI455X (gfx1250, wave32, WMMA).
// Dense GEMMs (LSTM x/h projections, CTC head, attention key projections) run
// on v_wmma_f32_16x16x32_f16 with 2/4-N-tile register blocking (A-frag reuse).
// Recurrent scans are persistent-workgroup kernels with h/c state in LDS and
// global_prefetch of the next timestep's xproj slab. Decoder (sequential over
// T and batch) is a single persistent workgroup doing VALU matvecs.
// ---------------------------------------------------------------------------

typedef __attribute__((ext_vector_type(16))) _Float16 v16h;
typedef __attribute__((ext_vector_type(8)))  _Float16 v8h;
typedef __attribute__((ext_vector_type(8)))  float    v8f;

#define SS   127   // encoder sequence length
#define BB   64    // batch
#define HH   128   // hidden
#define TT   100   // decode steps
#define VOC  80    // vocab
#define ROWS (SS*BB)   // 8128 rows, = 508 * 16

__device__ __forceinline__ float leakyf(float x)   { return x >= 0.f ? x : 0.01f * x; }
__device__ __forceinline__ float sigmf(float x)    { return 1.f / (1.f + expf(-x)); }

// ---------------------------------------------------------------------------
// Small utility kernels
// ---------------------------------------------------------------------------
__global__ void k_f32_to_f16_pad(const float* __restrict__ src, _Float16* __restrict__ dst,
                                 int rows, int cols, int dstRows) {
  int i = blockIdx.x * blockDim.x + threadIdx.x;
  int n = dstRows * cols;
  if (i >= n) return;
  int r = i / cols;
  dst[i] = (r < rows) ? (_Float16)src[i] : (_Float16)0.f;
}

__global__ void k_bias_sum(const float* __restrict__ a, const float* __restrict__ b,
                           float* __restrict__ dst, int n) {
  int i = blockIdx.x * blockDim.x + threadIdx.x;
  if (i < n) dst[i] = a[i] + b[i];
}

__global__ void k_bias_pad(const float* __restrict__ src, float* __restrict__ dst, int n, int nd) {
  int i = blockIdx.x * blockDim.x + threadIdx.x;
  if (i < nd) dst[i] = (i < n) ? src[i] : 0.f;
}

// ---------------------------------------------------------------------------
// CNN encoder
// ---------------------------------------------------------------------------
// conv1: x(64,3,64,1024) k(6,4) s(4,2) -> (64,8,15,511), leaky
__global__ void k_conv1(const float* __restrict__ x, const float* __restrict__ w,
                        const float* __restrict__ bias, float* __restrict__ out) {
  int i = blockIdx.x * blockDim.x + threadIdx.x;
  const int n = 64 * 8 * 15 * 511;
  if (i >= n) return;
  int wo = i % 511; int t = i / 511;
  int ho = t % 15;  t /= 15;
  int co = t % 8;   int b = t / 8;
  float acc = bias[co];
  for (int ci = 0; ci < 3; ci++)
    for (int kh = 0; kh < 6; kh++)
      for (int kw = 0; kw < 4; kw++)
        acc += x[((size_t)(b * 3 + ci) * 64 + ho * 4 + kh) * 1024 + wo * 2 + kw] *
               w[((co * 3 + ci) * 6 + kh) * 4 + kw];
  out[i] = leakyf(acc);
}

// conv2: (64,8,15,511) k(6,4) s1 -> (64,32,10,508), leaky
__global__ void k_conv2(const float* __restrict__ in, const float* __restrict__ w,
                        const float* __restrict__ bias, float* __restrict__ out) {
  int i = blockIdx.x * blockDim.x + threadIdx.x;
  const int n = 64 * 32 * 10 * 508;
  if (i >= n) return;
  int wo = i % 508; int t = i / 508;
  int ho = t % 10;  t /= 10;
  int co = t % 32;  int b = t / 32;
  float acc = bias[co];
  for (int ci = 0; ci < 8; ci++)
    for (int kh = 0; kh < 6; kh++)
      for (int kw = 0; kw < 4; kw++)
        acc += in[((size_t)(b * 8 + ci) * 15 + ho + kh) * 511 + wo + kw] *
               w[((co * 8 + ci) * 6 + kh) * 4 + kw];
  out[i] = leakyf(acc);
}

// pool1: (64,32,10,508) k(4,2) s(4,2) -> (64,32,2,254)
__global__ void k_pool1(const float* __restrict__ in, float* __restrict__ out) {
  int i = blockIdx.x * blockDim.x + threadIdx.x;
  const int n = 64 * 32 * 2 * 254;
  if (i >= n) return;
  int wo = i % 254; int t = i / 254;
  int ho = t % 2;   t /= 2;
  int c  = t % 32;  int b = t / 32;
  float m = -3.4e38f;
  for (int kh = 0; kh < 4; kh++)
    for (int kw = 0; kw < 2; kw++)
      m = fmaxf(m, in[((size_t)(b * 32 + c) * 10 + ho * 4 + kh) * 508 + wo * 2 + kw]);
  out[i] = m;
}

// conv3: (64,32,2,254) k3 pad1 -> (64,64,2,254), leaky
__global__ void k_conv3(const float* __restrict__ in, const float* __restrict__ w,
                        const float* __restrict__ bias, float* __restrict__ out) {
  int i = blockIdx.x * blockDim.x + threadIdx.x;
  const int n = 64 * 64 * 2 * 254;
  if (i >= n) return;
  int wo = i % 254; int t = i / 254;
  int ho = t % 2;   t /= 2;
  int co = t % 64;  int b = t / 64;
  float acc = bias[co];
  for (int ci = 0; ci < 32; ci++)
    for (int kh = 0; kh < 3; kh++) {
      int ih = ho + kh - 1;
      if (ih < 0 || ih >= 2) continue;
      for (int kw = 0; kw < 3; kw++) {
        int iw = wo + kw - 1;
        if (iw < 0 || iw >= 254) continue;
        acc += in[((size_t)(b * 32 + ci) * 2 + ih) * 254 + iw] *
               w[((co * 32 + ci) * 3 + kh) * 3 + kw];
      }
    }
  out[i] = leakyf(acc);
}

// pool2 (1,2) + reshape/permute -> xs0 f16 (S,B,128): f = c*2 + h
__global__ void k_pool2_pack(const float* __restrict__ c3, _Float16* __restrict__ xs0h) {
  int i = blockIdx.x * blockDim.x + threadIdx.x;
  const int n = SS * BB * 128;
  if (i >= n) return;
  int f = i % 128; int t = i / 128;
  int b = t % BB;  int s = t / BB;
  int c = f >> 1, h = f & 1;
  const float* p = c3 + ((size_t)(b * 64 + c) * 2 + h) * 254 + 2 * s;
  xs0h[((size_t)s * BB + b) * 128 + f] = (_Float16)fmaxf(p[0], p[1]);
}

// ---------------------------------------------------------------------------
// Generic WMMA GEMM:  C[M,N] f32 = A_f16[M,K] @ W_f16[N,K]^T + bias[N]
// M mult of 16, N mult of 16*NTW, K mult of 32. Each wave owns one M-tile and
// NTW adjacent N-tiles: the A fragment is loaded once per k-step and reused
// across NTW WMMAs. Fragment layouts per CDNA5 ISA §7.12.2.
// ---------------------------------------------------------------------------
template <int NTW>
__global__ __launch_bounds__(128) void k_gemm_wmma(
    const _Float16* __restrict__ A, const _Float16* __restrict__ W,
    const float* __restrict__ bias, float* __restrict__ C, _Float16* __restrict__ C16,
    int M, int N, int K, int totalGroups) {
  int wave = threadIdx.x >> 5;
  int lane = threadIdx.x & 31;
  int grp = blockIdx.x * 4 + wave;
  if (grp >= totalGroups) return;            // wave-uniform exit: EXEC stays all-ones
  int ngn = N / (16 * NTW);
  int mt = grp / ngn, ng = grp - mt * ngn;
  int hi = lane >> 4;
  int lr = lane & 15;
  const _Float16* arow  = A + (size_t)(mt * 16 + lr) * K;
  const _Float16* wrow0 = W + (size_t)(ng * 16 * NTW + lr) * K;
  v8f acc[NTW];
#pragma unroll
  for (int u = 0; u < NTW; u++) { v8f z = {}; acc[u] = z; }
  for (int k0 = 0; k0 < K; k0 += 32) {
    v8h al = *(const v8h*)(arow + k0 + hi * 8);        // A lane rows: K = hi*8 .. +7
    v8h ah = *(const v8h*)(arow + k0 + 16 + hi * 8);   //              K = 16+hi*8 .. +7
    v16h a;
#pragma unroll
    for (int q = 0; q < 8; q++) { a[q] = al[q]; a[q + 8] = ah[q]; }
#pragma unroll
    for (int u = 0; u < NTW; u++) {
      const _Float16* wrow = wrow0 + (size_t)u * 16 * K;
      v8h bl = *(const v8h*)(wrow + k0 + hi * 8);      // B(k,n)=W[n,k]: K = hi*8 .. +7
      v8h bh = *(const v8h*)(wrow + k0 + 16 + hi * 8); //               K = 16+hi*8 .. +7
      v16h b;
#pragma unroll
      for (int q = 0; q < 8; q++) { b[q] = bl[q]; b[q + 8] = bh[q]; }
      acc[u] = __builtin_amdgcn_wmma_f32_16x16x32_f16(false, a, false, b, (short)0, acc[u], false, false);
    }
  }
#pragma unroll
  for (int u = 0; u < NTW; u++) {
    int nn = ng * 16 * NTW + u * 16 + lr;
    float bv = bias ? bias[nn] : 0.f;
#pragma unroll
    for (int r = 0; r < 8; r++) {            // D: VGPR r -> M = hi*8+r, N = lane&15
      int mm = mt * 16 + hi * 8 + r;
      float v = acc[u][r] + bv;
      C[(size_t)mm * N + nn] = v;
      if (C16) C16[(size_t)mm * N + nn] = (_Float16)v;
    }
  }
}

// ---------------------------------------------------------------------------
// Bidirectional LSTM recurrent scan: one workgroup per direction.
// h (f16) and c (f32) live in LDS; per step, 16 waves compute the 64x512
// hidden projection with WMMA (A from LDS, reused across 4 N-tiles; W_hh from
// L2-resident global). The next step's xproj slab is prefetched during the
// GEMM. xproj already holds X@W_ih^T + (b_ih+b_hh).
// ---------------------------------------------------------------------------
__global__ __launch_bounds__(512) void k_lstm_scan(
    const _Float16* __restrict__ whh16,     // [2][512*128] f16
    const float*    __restrict__ xproj,     // [2][S*B*512] f32
    _Float16*       __restrict__ out_cat,   // (S,B,256) f16 : fwd -> [0,128), bwd -> [128,256)
    float*          __restrict__ gates_ws)  // [2][64*512] f32 scratch
{
  int dir = blockIdx.x;
  const _Float16* W = whh16 + (size_t)dir * 512 * 128;
  const float* XP   = xproj + (size_t)dir * ROWS * 512;
  float* G          = gates_ws + (size_t)dir * BB * 512;

  __shared__ __align__(16) _Float16 hbuf[BB * HH];   // 16 KB
  __shared__ float cbuf[BB * HH];                     // 32 KB
  for (int i = threadIdx.x; i < BB * HH; i += 512) { hbuf[i] = (_Float16)0.f; cbuf[i] = 0.f; }
  __syncthreads();

  int wave = threadIdx.x >> 5, lane = threadIdx.x & 31;
  int hi = lane >> 4, lr = lane & 15;

  for (int step = 0; step < SS; step++) {
    int t = dir ? (SS - 1 - step) : step;
    const float* xp_t = XP + (size_t)t * BB * 512;
    // prefetch next step's xproj slab (128 KB stream) while this GEMM runs
    if (step + 1 < SS) {
      int tn = dir ? (SS - 2 - step) : (step + 1);
      const float* xp_n = XP + (size_t)tn * BB * 512;
      for (int p = threadIdx.x * 16; p < BB * 512; p += 512 * 16)
        __builtin_prefetch(xp_n + p, 0, 1);
    }
    // phase 1: 64x512 output = 4 mt x 8 groups of 4 N-tiles; 2 groups per wave
    for (int g = wave; g < 32; g += 16) {
      int mt = g >> 3, ntg = g & 7;
      const _Float16* arow  = hbuf + (mt * 16 + lr) * HH;
      const _Float16* wrow0 = W + (size_t)(ntg * 64 + lr) * HH;
      v8f acc[4];
#pragma unroll
      for (int u = 0; u < 4; u++) { v8f z = {}; acc[u] = z; }
#pragma unroll
      for (int k0 = 0; k0 < HH; k0 += 32) {
        v8h al = *(const v8h*)(arow + k0 + hi * 8);
        v8h ah = *(const v8h*)(arow + k0 + 16 + hi * 8);
        v16h a;
#pragma unroll
        for (int q = 0; q < 8; q++) { a[q] = al[q]; a[q + 8] = ah[q]; }
#pragma unroll
        for (int u = 0; u < 4; u++) {
          const _Float16* wrow = wrow0 + (size_t)u * 16 * HH;
          v8h bl = *(const v8h*)(wrow + k0 + hi * 8);
          v8h bh = *(const v8h*)(wrow + k0 + 16 + hi * 8);
          v16h b;
#pragma unroll
          for (int q = 0; q < 8; q++) { b[q] = bl[q]; b[q + 8] = bh[q]; }
          acc[u] = __builtin_amdgcn_wmma_f32_16x16x32_f16(false, a, false, b, (short)0, acc[u], false, false);
        }
      }
#pragma unroll
      for (int u = 0; u < 4; u++) {
        int nn = ntg * 64 + u * 16 + lr;
#pragma unroll
        for (int r = 0; r < 8; r++) {
          int mm = mt * 16 + hi * 8 + r;
          G[mm * 512 + nn] = acc[u][r] + xp_t[mm * 512 + nn];
        }
      }
    }
    __threadfence_block();
    __syncthreads();
    // phase 2: pointwise gate math, h/c update
    for (int e = threadIdx.x; e < BB * HH; e += 512) {
      int b = e >> 7, j = e & 127;
      float gi = G[b * 512 + j];
      float gf = G[b * 512 + 128 + j];
      float gg = G[b * 512 + 256 + j];
      float go = G[b * 512 + 384 + j];
      float c = sigmf(gf) * cbuf[e] + sigmf(gi) * tanhf(gg);
      float h = sigmf(go) * tanhf(c);
      cbuf[e] = c;
      hbuf[e] = (_Float16)h;
      out_cat[((size_t)t * BB + b) * 256 + dir * HH + j] = (_Float16)h;
    }
    __syncthreads();
  }
}

// ---------------------------------------------------------------------------
// CTC log-softmax over 81 classes (logits padded to 96 cols)
// ---------------------------------------------------------------------------
__global__ __launch_bounds__(128) void k_logsoftmax81(const float* __restrict__ lg,
                                                      float* __restrict__ out) {
  __shared__ float red[128];
  int row = blockIdx.x, tid = threadIdx.x;
  float v = (tid < 81) ? lg[(size_t)row * 96 + tid] : -3.4e38f;
  red[tid] = v; __syncthreads();
  for (int s = 64; s > 0; s >>= 1) { if (tid < s) red[tid] = fmaxf(red[tid], red[tid + s]); __syncthreads(); }
  float mx = red[0]; __syncthreads();
  float e = (tid < 81) ? expf(v - mx) : 0.f;
  red[tid] = e; __syncthreads();
  for (int s = 64; s > 0; s >>= 1) { if (tid < s) red[tid] += red[tid + s]; __syncthreads(); }
  float lse = logf(red[0]) + mx;
  if (tid < 81) out[(size_t)row * 81 + tid] = v - lse;
}

// ---------------------------------------------------------------------------
// Attention decoder: single persistent workgroup. Faithful to the reference's
// single shared (h,c) state scanned over T then over batch.
// ---------------------------------------------------------------------------
__global__ __launch_bounds__(512) void k_decoder(
    const float* __restrict__ ks,    // (S,B,128) = W_s(keys)+b_s
    const float* __restrict__ keys,  // (S,B,128)
    const float* __restrict__ attWh, const float* __restrict__ attv,
    const float* __restrict__ emb,   const int* __restrict__ y,
    const float* __restrict__ wih,   const float* __restrict__ whh,
    const float* __restrict__ bih,   const float* __restrict__ bhh,
    const float* __restrict__ outw,  const float* __restrict__ outb,
    float* __restrict__ ctx_ws,      // (64,128) f32 scratch
    float* __restrict__ out)         // (B,T,80)
{
  __shared__ float sh_scores[BB * SS];  // 32.5 KB
  __shared__ float sh_h[HH], sh_c[HH], sh_q[HH];
  __shared__ float sh_ins[192];
  __shared__ float sh_g[512];
  int tid = threadIdx.x;
  if (tid < HH) { sh_h[tid] = 0.f; sh_c[tid] = 0.f; }
  __syncthreads();

  for (int t = 0; t < TT; t++) {
    // q = h @ Wh^T
    if (tid < HH) {
      float a = 0.f;
      for (int k = 0; k < HH; k++) a += sh_h[k] * attWh[tid * HH + k];
      sh_q[tid] = a;
    }
    __syncthreads();
    // scores[b,s] = tanh(q + ks[b,s,:]) . v
    for (int e = tid; e < BB * SS; e += 512) {
      int b = e / SS, s = e - b * SS;
      const float* kr = ks + ((size_t)s * BB + b) * HH;
      float a = 0.f;
      for (int j = 0; j < HH; j++) a += tanhf(sh_q[j] + kr[j]) * attv[j];
      sh_scores[e] = a;
    }
    __syncthreads();
    // softmax over s for each b
    if (tid < BB) {
      float mx = -3.4e38f;
      for (int s = 0; s < SS; s++) mx = fmaxf(mx, sh_scores[tid * SS + s]);
      float sum = 0.f;
      for (int s = 0; s < SS; s++) { float e = expf(sh_scores[tid * SS + s] - mx); sh_scores[tid * SS + s] = e; sum += e; }
      float inv = 1.f / sum;
      for (int s = 0; s < SS; s++) sh_scores[tid * SS + s] *= inv;
    }
    __syncthreads();
    // ctx[b,j] = sum_s w[b,s] * keys[b,s,j]
    for (int e = tid; e < BB * HH; e += 512) {
      int b = e >> 7, j = e & 127;
      float a = 0.f;
      for (int s = 0; s < SS; s++) a += sh_scores[b * SS + s] * keys[((size_t)s * BB + b) * HH + j];
      ctx_ws[e] = a;
    }
    __threadfence_block();
    __syncthreads();
    // inner sequential scan over batch with the single shared (h,c)
    for (int b = 0; b < BB; b++) {
      if (tid < 192) {
        if (tid < 64) { int tok = y[b * TT + t]; sh_ins[tid] = emb[tok * 64 + tid]; }
        else          sh_ins[tid] = ctx_ws[b * HH + (tid - 64)];
      }
      __syncthreads();
      {
        float a = bih[tid] + bhh[tid];
        const float* wr = wih + (size_t)tid * 192;
        for (int k = 0; k < 192; k++) a += wr[k] * sh_ins[k];
        const float* hr = whh + (size_t)tid * HH;
        for (int k = 0; k < HH; k++) a += hr[k] * sh_h[k];
        sh_g[tid] = a;
      }
      __syncthreads();
      if (tid < HH) {
        float gi = sh_g[tid], gf = sh_g[128 + tid], gg = sh_g[256 + tid], go = sh_g[384 + tid];
        float c = sigmf(gf) * sh_c[tid] + sigmf(gi) * tanhf(gg);
        sh_c[tid] = c;
        sh_h[tid] = sigmf(go) * tanhf(c);
      }
      __syncthreads();
      if (tid < VOC) {
        float a = outb[tid];
        const float* wr = outw + (size_t)tid * HH;
        for (int j = 0; j < HH; j++) a += wr[j] * sh_h[j];
        out[((size_t)b * TT + t) * VOC + tid] = a;
      }
      __syncthreads();
    }
  }
}

// ---------------------------------------------------------------------------
// Host orchestration
// ---------------------------------------------------------------------------
static inline size_t alignup(size_t x) { return (x + 255) & ~(size_t)255; }

extern "C" void kernel_launch(void* const* d_in, const int* in_sizes, int n_in,
                              void* d_out, int out_size, void* d_ws, size_t ws_size,
                              hipStream_t stream) {
  // ---- input leaf mapping (setup_inputs() insertion order, depth-first) ----
  // 0:x 1:y 2..7:conv(w1,b1,w2,b2,w3,b3)
  // 8 + (l*2+d)*4 + {0:w_ih,1:w_hh,2:b_ih,3:b_hh}  for l in 0..2, d in 0..1
  // 32:ctc_w 33:ctc_b 34:red_w 35:red_b 36:att_Wh 37:att_Ws 38:att_bs 39:att_v
  // 40:emb 41:dec_wih 42:dec_whh 43:dec_bih 44:dec_bhh 45:out_w 46:out_b
  const float* x     = (const float*)d_in[0];
  const int*   y     = (const int*)d_in[1];
  const float* w1    = (const float*)d_in[2];
  const float* b1    = (const float*)d_in[3];
  const float* w2    = (const float*)d_in[4];
  const float* b2    = (const float*)d_in[5];
  const float* w3    = (const float*)d_in[6];
  const float* b3    = (const float*)d_in[7];
  const float* ctc_w = (const float*)d_in[32];
  const float* ctc_b = (const float*)d_in[33];
  const float* red_w = (const float*)d_in[34];
  const float* red_b = (const float*)d_in[35];
  const float* attWh = (const float*)d_in[36];
  const float* attWs = (const float*)d_in[37];
  const float* attbs = (const float*)d_in[38];
  const float* attv  = (const float*)d_in[39];
  const float* emb   = (const float*)d_in[40];
  const float* dwih  = (const float*)d_in[41];
  const float* dwhh  = (const float*)d_in[42];
  const float* dbih  = (const float*)d_in[43];
  const float* dbhh  = (const float*)d_in[44];
  const float* outw  = (const float*)d_in[45];
  const float* outb  = (const float*)d_in[46];

  // ---- workspace layout (~125 MB) ----
  char* ws = (char*)d_ws;
  size_t off = 0;
  auto take = [&](size_t bytes) { void* p = ws + off; off += alignup(bytes); return p; };
  float*    c1buf   = (float*)   take((size_t)64 * 8 * 15 * 511 * 4);
  float*    c2buf   = (float*)   take((size_t)64 * 32 * 10 * 508 * 4);
  float*    p1buf   = (float*)   take((size_t)64 * 32 * 2 * 254 * 4);
  float*    c3buf   = (float*)   take((size_t)64 * 64 * 2 * 254 * 4);
  _Float16* xs0h    = (_Float16*)take((size_t)ROWS * 128 * 2);
  _Float16* catA    = (_Float16*)take((size_t)ROWS * 256 * 2);
  _Float16* catB    = (_Float16*)take((size_t)ROWS * 256 * 2);
  float*    xproj   = (float*)   take((size_t)2 * ROWS * 512 * 4);
  float*    gws     = (float*)   take((size_t)2 * BB * 512 * 4);
  _Float16* wih16   = (_Float16*)take((size_t)2 * 512 * 256 * 2);
  _Float16* whh16   = (_Float16*)take((size_t)2 * 512 * 128 * 2);
  float*    bsum    = (float*)   take((size_t)2 * 512 * 4);
  _Float16* ctcw16  = (_Float16*)take((size_t)96 * 256 * 2);
  float*    ctcb96  = (float*)   take((size_t)96 * 4);
  _Float16* redw16  = (_Float16*)take((size_t)128 * 256 * 2);
  _Float16* attws16 = (_Float16*)take((size_t)128 * 128 * 2);
  float*    lg96    = (float*)   take((size_t)ROWS * 96 * 4);
  float*    keysb   = (float*)   take((size_t)ROWS * 128 * 4);
  _Float16* keys16  = (_Float16*)take((size_t)ROWS * 128 * 2);
  float*    ksbuf   = (float*)   take((size_t)ROWS * 128 * 4);
  float*    ctxws   = (float*)   take((size_t)BB * HH * 4);
  (void)ws_size; (void)in_sizes; (void)n_in; (void)out_size;

  float* out_ctc = (float*)d_out;                         // (S,B,81)
  float* out_dec = (float*)d_out + (size_t)SS * BB * 81;  // (B,T,80)

  // ---- CNN encoder ----
  { int n = 64 * 8 * 15 * 511;   k_conv1<<<(n + 255) / 256, 256, 0, stream>>>(x, w1, b1, c1buf); }
  { int n = 64 * 32 * 10 * 508;  k_conv2<<<(n + 255) / 256, 256, 0, stream>>>(c1buf, w2, b2, c2buf); }
  { int n = 64 * 32 * 2 * 254;   k_pool1<<<(n + 255) / 256, 256, 0, stream>>>(c2buf, p1buf); }
  { int n = 64 * 64 * 2 * 254;   k_conv3<<<(n + 255) / 256, 256, 0, stream>>>(p1buf, w3, b3, c3buf); }
  { int n = ROWS * 128;          k_pool2_pack<<<(n + 255) / 256, 256, 0, stream>>>(c3buf, xs0h); }

  // ---- 3-layer bidirectional LSTM ----
  for (int l = 0; l < 3; l++) {
    int Din = (l == 0) ? 128 : 256;
    const _Float16* A = (l == 0) ? xs0h : ((l == 1) ? catA : catB);
    _Float16* catOut  = (l == 0) ? catA : ((l == 1) ? catB : catA);
    for (int d = 0; d < 2; d++) {
      int base = 8 + (l * 2 + d) * 4;
      { int n = 512 * Din; k_f32_to_f16_pad<<<(n + 255) / 256, 256, 0, stream>>>(
            (const float*)d_in[base + 0], wih16 + (size_t)d * 512 * 256, 512, Din, 512); }
      { int n = 512 * 128; k_f32_to_f16_pad<<<(n + 255) / 256, 256, 0, stream>>>(
            (const float*)d_in[base + 1], whh16 + (size_t)d * 512 * 128, 512, 128, 512); }
      k_bias_sum<<<2, 256, 0, stream>>>((const float*)d_in[base + 2],
                                        (const float*)d_in[base + 3], bsum + d * 512, 512);
      int groups = (ROWS / 16) * (512 / 64);   // 4064
      k_gemm_wmma<4><<<groups / 4, 128, 0, stream>>>(
          A, wih16 + (size_t)d * 512 * 256, bsum + d * 512,
          xproj + (size_t)d * ROWS * 512, nullptr, ROWS, 512, Din, groups);
    }
    k_lstm_scan<<<2, 512, 0, stream>>>(whh16, xproj, catOut, gws);
  }
  const _Float16* enc16 = catA;  // (S,B,256) f16

  // ---- CTC head ----
  { int n = 96 * 256; k_f32_to_f16_pad<<<(n + 255) / 256, 256, 0, stream>>>(ctc_w, ctcw16, 81, 256, 96); }
  k_bias_pad<<<1, 96, 0, stream>>>(ctc_b, ctcb96, 81, 96);
  { int groups = (ROWS / 16) * (96 / 32);   // 1524
    k_gemm_wmma<2><<<groups / 4, 128, 0, stream>>>(enc16, ctcw16, ctcb96, lg96, nullptr, ROWS, 96, 256, groups); }
  k_logsoftmax81<<<ROWS, 128, 0, stream>>>(lg96, out_ctc);

  // ---- keys = enc @ red_w^T + red_b ; ks = keys @ att_Ws^T + att_bs ----
  { int n = 128 * 256; k_f32_to_f16_pad<<<(n + 255) / 256, 256, 0, stream>>>(red_w, redw16, 128, 256, 128); }
  { int n = 128 * 128; k_f32_to_f16_pad<<<(n + 255) / 256, 256, 0, stream>>>(attWs, attws16, 128, 128, 128); }
  { int groups = (ROWS / 16) * (128 / 64);  // 1016
    k_gemm_wmma<4><<<groups / 4, 128, 0, stream>>>(enc16, redw16, red_b, keysb, keys16, ROWS, 128, 256, groups); }
  { int groups = (ROWS / 16) * (128 / 64);  // 1016
    k_gemm_wmma<4><<<groups / 4, 128, 0, stream>>>(keys16, attws16, attbs, ksbuf, nullptr, ROWS, 128, 128, groups); }

  // ---- attention decoder ----
  k_decoder<<<1, 512, 0, stream>>>(ksbuf, keysb, attWh, attv, emb, y,
                                   dwih, dwhh, dbih, dbhh, outw, outb, ctxws, out_dec);
}
